// Attention_927712935992
// MI455X (gfx1250) — matrix operand order repaired and checked by reading, not compiled, not run on hardware
//
#include <hip/hip_runtime.h>
#include <math.h>

typedef _Float16 f16;
typedef __attribute__((ext_vector_type(16))) _Float16 v16h;
typedef __attribute__((ext_vector_type(8)))  _Float16 v8h;
typedef __attribute__((ext_vector_type(8)))  float    v8f;

constexpr int kL   = 2048;   // sequence length
constexpr int kB   = 2;      // batch
constexpr int kD   = 1024;   // model dim
constexpr int kH   = 16;     // heads
constexpr int kHD  = 64;     // head dim
constexpr int kM   = kL * kB;     // 4096 token rows
constexpr int kNQKV = 3 * kD;     // 3072

#define DEV __device__ __forceinline__

DEV v8f wmma16(v16h a, v16h b, v8f c) {
  // D = A(16x32 f16) x B(32x16 f16) + C(16x16 f32)
  return __builtin_amdgcn_wmma_f32_16x16x32_f16(
      /*neg_a=*/false, a, /*neg_b=*/false, b,
      /*c_mod=*/(short)0, c, /*reuse_a=*/false, /*reuse_b=*/false);
}

// A fragment (16x32, 16-bit): lane&15 = M row; element e -> K = 8*(lane>>4) + 16*(e>>3) + (e&7)
// => per lane: two contiguous runs of 8 halves at K = kb and K = kb+16
DEV v16h ld_frag_a(const f16* base, int stride, int lane) {
  const int r  = lane & 15;
  const int kb = (lane >> 4) * 8;
  const f16* p = base + (size_t)r * stride + kb;
  v8h lo = *(const v8h*)(p);
  v8h hi = *(const v8h*)(p + 16);
  v16h o;
#pragma unroll
  for (int i = 0; i < 8; ++i) { o[i] = lo[i]; o[i + 8] = hi[i]; }
  return o;
}

// B fragment (32x16, 16-bit): lane&15 = N col; element e -> K = 8*(lane>>4) + 16*(e>>3) + (e&7)
// => per lane: two contiguous runs of 8 halves at K = kb and K = kb+16
DEV v16h ld_frag_b(const f16* base, int stride, int lane) {
  const int r  = lane & 15;
  const int kb = (lane >> 4) * 8;
  const f16* p = base + (size_t)r * stride + kb;
  v8h lo = *(const v8h*)(p);
  v8h hi = *(const v8h*)(p + 16);
  v16h o;
#pragma unroll
  for (int i = 0; i < 8; ++i) { o[i] = lo[i]; o[i + 8] = hi[i]; }
  return o;
}

// ---------------- LayerNorm (fp32) -> xn (f16) ----------------
__global__ void ln_kernel(const float* __restrict__ x, const float* __restrict__ w,
                          const float* __restrict__ b, f16* __restrict__ xn) {
  const int row = blockIdx.x;
  const int tid = threadIdx.x;        // 256 threads, 4 elems each
  const float* xr = x + (size_t)row * kD;
  __shared__ float red[256];
  float4 v = ((const float4*)xr)[tid];
  red[tid] = v.x + v.y + v.z + v.w;
  __syncthreads();
  for (int off = 128; off > 0; off >>= 1) {
    if (tid < off) red[tid] += red[tid + off];
    __syncthreads();
  }
  const float mu = red[0] * (1.0f / kD);
  __syncthreads();
  const float d0 = v.x - mu, d1 = v.y - mu, d2 = v.z - mu, d3 = v.w - mu;
  red[tid] = d0 * d0 + d1 * d1 + d2 * d2 + d3 * d3;
  __syncthreads();
  for (int off = 128; off > 0; off >>= 1) {
    if (tid < off) red[tid] += red[tid + off];
    __syncthreads();
  }
  const float var = red[0] * (1.0f / kD);
  const float inv = rsqrtf(var + 1e-12f);
  const int c = tid * 4;
  f16* dst = xn + (size_t)row * kD + c;
  dst[0] = (f16)(d0 * inv * w[c + 0] + b[c + 0]);
  dst[1] = (f16)(d1 * inv * w[c + 1] + b[c + 1]);
  dst[2] = (f16)(d2 * inv * w[c + 2] + b[c + 2]);
  dst[3] = (f16)(d3 * inv * w[c + 3] + b[c + 3]);
}

// ---------------- fp32 -> f16 weight conversion ----------------
__global__ void cvt_kernel(const float* __restrict__ src, f16* __restrict__ dst, int n) {
  const int i = blockIdx.x * blockDim.x + threadIdx.x;
  if (i < n) dst[i] = (f16)src[i];
}

// ---------------- QKV GEMM: C[4096,3072] = xn @ qkv_w^T + b; scatter to Q,K,V^T ----------------
// Register-tiled: each wave computes 32(M) x 64(N): 2 A frags x 4 B frags -> 8 WMMA / k-step.
__global__ void gemm_qkv_kernel(const f16* __restrict__ xn, const f16* __restrict__ wf,
                                const float* __restrict__ bias,
                                f16* __restrict__ qf, f16* __restrict__ kf,
                                f16* __restrict__ vtf) {
  const int lane = threadIdx.x & 31;
  const int wave = threadIdx.x >> 5;
  const int mt = blockIdx.x;               // 32-row tile of tokens
  const int ng = blockIdx.y * 4 + wave;    // 64-col group of QKV outputs
  const f16* a0base = xn + (size_t)mt * 32 * kD;
  const f16* a1base = a0base + 16 * kD;
  const f16* bbase  = wf + (size_t)ng * 64 * kD;  // W row n contiguous in k
  v8f acc[2][4] = {};
#pragma unroll 2
  for (int k0 = 0; k0 < kD; k0 += 32) {
    const v16h a0 = ld_frag_a(a0base + k0, kD, lane);
    const v16h a1 = ld_frag_a(a1base + k0, kD, lane);
#pragma unroll
    for (int t = 0; t < 4; ++t) {
      const v16h b = ld_frag_b(bbase + (size_t)t * 16 * kD + k0, kD, lane);
      acc[0][t] = wmma16(a0, b, acc[0][t]);
      acc[1][t] = wmma16(a1, b, acc[1][t]);
    }
  }
  // C layout: lane&15 = n, VGPR j -> m = j + 8*(lane>>4)
  const int nl = lane & 15, hig = lane >> 4;
#pragma unroll
  for (int t = 0; t < 4; ++t) {
    const int n = ng * 64 + t * 16 + nl;
    const float bv = bias[n];
    const int which = n >> 10;          // 0=Q 1=K 2=V
    const int dcol = n & (kD - 1);
    const int h   = dcol >> 6;
    const int hdi = dcol & (kHD - 1);
#pragma unroll
    for (int sub = 0; sub < 2; ++sub) {
#pragma unroll
      for (int j = 0; j < 8; ++j) {
        const int m = mt * 32 + sub * 16 + j + hig * 8;   // token row = l*B + b
        const int l  = m >> 1;
        const int bb = m & 1;
        const int bh = bb * kH + h;
        const f16 val = (f16)(acc[sub][t][j] + bv);
        if (which == 0)      qf [((size_t)bh * kL + l) * kHD + hdi] = val;  // (BH, L, hd)
        else if (which == 1) kf [((size_t)bh * kL + l) * kHD + hdi] = val;  // (BH, L, hd)
        else                 vtf[((size_t)bh * kHD + hdi) * kL + l] = val;  // (BH, hd, L)
      }
    }
  }
}

// ---------------- Flash attention: per (head, 64-query block), 4 waves x 16 queries ----------------
__global__ void attn_kernel(const f16* __restrict__ qf, const f16* __restrict__ kf,
                            const f16* __restrict__ vtf, const int* __restrict__ amask,
                            f16* __restrict__ ctx) {
  const int lane = threadIdx.x & 31;
  const int wave = threadIdx.x >> 5;
  const int bh   = blockIdx.y;        // 0..31
  const int bidx = bh >> 4;           // batch
  const int h    = bh & (kH - 1);
  const int q0   = blockIdx.x * 64 + wave * 16;
  const int nl = lane & 15, hig = lane >> 4;

  __shared__ __align__(16) f16 lds_p[4][16 * 32];   // per-wave P tile (16 q x 32 keys)
  f16* pl = &lds_p[wave][0];

  const f16* qbase = qf + ((size_t)bh * kL + q0) * kHD;
  const v16h qa0 = ld_frag_a(qbase,      kHD, lane);   // hd 0..31
  const v16h qa1 = ld_frag_a(qbase + 32, kHD, lane);   // hd 32..63

  v8f o0 = {}, o1 = {}, o2 = {}, o3 = {};   // 16 q x 64 hd accumulator
  float mrun[8], lrun[8];
#pragma unroll
  for (int j = 0; j < 8; ++j) { mrun[j] = -INFINITY; lrun[j] = 0.f; }

  const float scale = 0.125f;               // 1/sqrt(64)
  const int* mrow = amask + bidx * kL;
  const f16* kf_h = kf + (size_t)bh * kL * kHD;
  const f16* vt_h = vtf + (size_t)bh * kHD * kL;

  for (int kk = 0; kk < kL; kk += 32) {
    // prefetch next key/value block into L2 (global_prefetch_b8)
    if (kk + 32 < kL) {
      __builtin_prefetch(kf_h + (size_t)(kk + 32 + lane) * kHD, 0, 3);
      __builtin_prefetch(vt_h + (size_t)(lane * 2) * kL + kk + 32, 0, 3);
    }
    // S = Q @ K^T : two 16x16 score tiles over keys [kk..kk+31]
    const f16* kb0 = kf_h + (size_t)kk * kHD;
    const f16* kb1 = kb0 + 16 * kHD;
    v8f s0 = {}, s1 = {};
    s0 = wmma16(qa0, ld_frag_b(kb0,      kHD, lane), s0);
    s0 = wmma16(qa1, ld_frag_b(kb0 + 32, kHD, lane), s0);
    s1 = wmma16(qa0, ld_frag_b(kb1,      kHD, lane), s1);
    s1 = wmma16(qa1, ld_frag_b(kb1 + 32, kHD, lane), s1);

    const bool keep0 = mrow[kk + nl] != 0;        // key column owned by this lane, tile 0
    const bool keep1 = mrow[kk + 16 + nl] != 0;   // tile 1

    float alph[8];
#pragma unroll
    for (int j = 0; j < 8; ++j) {
      float a0 = keep0 ? s0[j] * scale : -INFINITY;
      float a1 = keep1 ? s1[j] * scale : -INFINITY;
      // row (m = j + 8*hig) lives in the 16 lanes of this half-wave
      float rm = fmaxf(a0, a1);
#pragma unroll
      for (int off = 1; off < 16; off <<= 1)
        rm = fmaxf(rm, __shfl_xor(rm, off, 16));
      const float mnew  = fmaxf(mrun[j], rm);
      const float alpha = (mrun[j] > -1e30f) ? __expf(mrun[j] - mnew) : 0.f;
      const float p0 = (a0 > -1e30f) ? __expf(a0 - mnew) : 0.f;
      const float p1 = (a1 > -1e30f) ? __expf(a1 - mnew) : 0.f;
      float rs = p0 + p1;
#pragma unroll
      for (int off = 1; off < 16; off <<= 1)
        rs += __shfl_xor(rs, off, 16);
      lrun[j] = alpha * lrun[j] + rs;
      mrun[j] = mnew;
      alph[j] = alpha;
      // stage P into LDS in A-layout source form (row-major 16x32)
      const int m = j + hig * 8;
      pl[m * 32 + nl]      = (f16)p0;
      pl[m * 32 + 16 + nl] = (f16)p1;
    }
#pragma unroll
    for (int j = 0; j < 8; ++j) {
      o0[j] *= alph[j]; o1[j] *= alph[j]; o2[j] *= alph[j]; o3[j] *= alph[j];
    }
    // O += P @ V   (same-wave DS ops are in-order: stores above complete before these loads)
    const v16h pa = ld_frag_a(pl, 32, lane);
    const f16* vb = vt_h + kk;                          // V^T: (hd, L)
    o0 = wmma16(pa, ld_frag_b(vb + 0 * 16 * kL, kL, lane), o0);
    o1 = wmma16(pa, ld_frag_b(vb + 1 * 16 * kL, kL, lane), o1);
    o2 = wmma16(pa, ld_frag_b(vb + 2 * 16 * kL, kL, lane), o2);
    o3 = wmma16(pa, ld_frag_b(vb + 3 * 16 * kL, kL, lane), o3);
  }

  // normalize and write context in (L, B, D) f16 row-major for the out-proj GEMM
#pragma unroll
  for (int j = 0; j < 8; ++j) {
    const int l = q0 + j + hig * 8;
    const float inv = lrun[j] > 0.f ? 1.0f / lrun[j] : 0.f;
    f16* dst = ctx + ((size_t)l * kB + bidx) * kD + h * kHD;
    dst[0 * 16 + nl] = (f16)(o0[j] * inv);
    dst[1 * 16 + nl] = (f16)(o1[j] * inv);
    dst[2 * 16 + nl] = (f16)(o2[j] * inv);
    dst[3 * 16 + nl] = (f16)(o3[j] * inv);
  }
}

// ---------------- Output GEMM: out[4096,1024] = ctx @ out_w^T + out_b (fp32 out) ----------------
// Register-tiled: each wave computes 32(M) x 64(N).
__global__ void gemm_out_kernel(const f16* __restrict__ ctx, const f16* __restrict__ wf,
                                const float* __restrict__ bias, float* __restrict__ out) {
  const int lane = threadIdx.x & 31;
  const int wave = threadIdx.x >> 5;
  const int mt = blockIdx.x;               // 32-row tile
  const int ng = blockIdx.y * 4 + wave;    // 64-col group
  const f16* a0base = ctx + (size_t)mt * 32 * kD;
  const f16* a1base = a0base + 16 * kD;
  const f16* bbase  = wf + (size_t)ng * 64 * kD;
  v8f acc[2][4] = {};
#pragma unroll 2
  for (int k0 = 0; k0 < kD; k0 += 32) {
    const v16h a0 = ld_frag_a(a0base + k0, kD, lane);
    const v16h a1 = ld_frag_a(a1base + k0, kD, lane);
#pragma unroll
    for (int t = 0; t < 4; ++t) {
      const v16h b = ld_frag_b(bbase + (size_t)t * 16 * kD + k0, kD, lane);
      acc[0][t] = wmma16(a0, b, acc[0][t]);
      acc[1][t] = wmma16(a1, b, acc[1][t]);
    }
  }
  const int nl = lane & 15, hig = lane >> 4;
#pragma unroll
  for (int t = 0; t < 4; ++t) {
    const int n = ng * 64 + t * 16 + nl;
    const float bv = bias[n];
#pragma unroll
    for (int sub = 0; sub < 2; ++sub) {
#pragma unroll
      for (int j = 0; j < 8; ++j) {
        const int m = mt * 32 + sub * 16 + j + hig * 8;
        out[(size_t)m * kD + n] = acc[sub][t][j] + bv;
      }
    }
  }
}

extern "C" void kernel_launch(void* const* d_in, const int* in_sizes, int n_in,
                              void* d_out, int out_size, void* d_ws, size_t ws_size,
                              hipStream_t stream) {
  (void)in_sizes; (void)n_in; (void)out_size; (void)ws_size;
  const float* x     = (const float*)d_in[0];
  const int*   amask = (const int*)  d_in[1];
  const float* ln_w  = (const float*)d_in[2];
  const float* ln_b  = (const float*)d_in[3];
  const float* qkv_w = (const float*)d_in[4];
  const float* qkv_b = (const float*)d_in[5];
  const float* out_w = (const float*)d_in[6];
  const float* out_b = (const float*)d_in[7];
  float* out = (float*)d_out;

  // workspace layout (all f16, 48 MiB total)
  char* ws = (char*)d_ws;
  constexpr size_t MB = 1024ull * 1024ull;
  f16* xnf   = (f16*)(ws + 0 * MB);    // 4096x1024      (8 MiB)
  f16* wqkvf = (f16*)(ws + 8 * MB);    // 3072x1024      (6 MiB)
  f16* wof   = (f16*)(ws + 14 * MB);   // 1024x1024      (2 MiB)
  f16* qf    = (f16*)(ws + 16 * MB);   // (32,2048,64)   (8 MiB)
  f16* kf    = (f16*)(ws + 24 * MB);   // (32,2048,64)   (8 MiB)
  f16* vtf   = (f16*)(ws + 32 * MB);   // (32,64,2048)   (8 MiB)
  f16* ctxf  = (f16*)(ws + 40 * MB);   // 4096x1024      (8 MiB)

  ln_kernel<<<kM, 256, 0, stream>>>(x, ln_w, ln_b, xnf);
  cvt_kernel<<<(kNQKV * kD + 255) / 256, 256, 0, stream>>>(qkv_w, wqkvf, kNQKV * kD);
  cvt_kernel<<<(kD * kD + 255) / 256, 256, 0, stream>>>(out_w, wof, kD * kD);
  gemm_qkv_kernel<<<dim3(kM / 32, kNQKV / 256), 128, 0, stream>>>(xnf, wqkvf, qkv_b,
                                                                  qf, kf, vtf);
  attn_kernel<<<dim3(kL / 64, kB * kH), 128, 0, stream>>>(qf, kf, vtf, amask, ctxf);
  gemm_out_kernel<<<dim3(kM / 32, kD / 256), 128, 0, stream>>>(ctxf, wof, out_b, out);
}
